// Wavenet_62036507623565
// MI455X (gfx1250) — matrix operand order repaired and checked by reading, not compiled, not run on hardware
//
#include <hip/hip_runtime.h>
#include <hip/hip_bf16.h>
#include <math.h>

// WaveNet on MI455X (gfx1250, wave32, WMMA).
// Key algorithmic reduction: the reference OVERWRITES `skip` each layer and
// only uses h[:, :, -1], so the output depends only on layer-15's skip at the
// final position -> a 77-position trailing window of the sequence. We keep an
// 80-wide window per batch in LDS and run all 16 layers in one kernel,
// using v_wmma_f32_16x16x32_f16 for every 32x32 conv tap (K=32 exactly).

typedef __attribute__((ext_vector_type(16))) _Float16 v16h;
typedef __attribute__((ext_vector_type(8)))  float    v8f;

#define GUARD 16
#define WCOLS 128   // 16 guard + 96 data + 16 slack
#define WIN   80    // computed window of positions (>= 77 needed)

// A fragment: 16x32 f16, A[M=oc0+m][K] ; per-lane K pattern from ISA table.
// weight element address: base[(oc)*ocStride + k*icStride]
__device__ __forceinline__ v16h a_frag(const float* base, int oc0, int ocStride,
                                       int icStride, int m, int hi) {
    v16h a;
#pragma unroll
    for (int e = 0; e < 16; ++e) {
        int k = (e < 8) ? (8 * hi + e) : (8 + 8 * hi + e); // e>=8: 16+8*hi+(e-8)
        a[e] = (_Float16)base[(oc0 + m) * ocStride + k * icStride];
    }
    return a;
}

// B fragment: 32x16 f16, B[K][N=t0+n+shift] from LDS buffer buf[ch][col].
// K = 8*hi + e (e<8), 16 + 8*hi + (e-8) (e>=8), matching the A fragment K order.
__device__ __forceinline__ v16h b_frag(const float (*buf)[WCOLS], int t0, int shift,
                                       int n, int hi) {
    v16h b;
    int col = GUARD + t0 + n + shift;  // shift >= -9, so col >= 7 always valid
#pragma unroll
    for (int e = 0; e < 16; ++e) {
        b[e] = (_Float16)buf[(e < 8) ? (8 * hi + e) : (8 + 8 * hi + e)][col];
    }
    return b;
}

__device__ __forceinline__ v8f wmma_f16(v16h a, v16h b, v8f c) {
    return __builtin_amdgcn_wmma_f32_16x16x32_f16(false, a, false, b,
                                                  (short)0, c, false, false);
}

__global__ __launch_bounds__(128)
void wavenet_gfx1250_kernel(const int* __restrict__ tokens,
                            const float* __restrict__ emb,
                            const float* __restrict__ init_w, const float* __restrict__ init_b,
                            const float* __restrict__ dil_w,  const float* __restrict__ dil_b,
                            const float* __restrict__ filt_w, const float* __restrict__ filt_b,
                            const float* __restrict__ gate_w, const float* __restrict__ gate_b,
                            const float* __restrict__ res_w,  const float* __restrict__ res_b,
                            const float* __restrict__ skip_w, const float* __restrict__ skip_b,
                            const float* __restrict__ end1_w, const float* __restrict__ end1_b,
                            const float* __restrict__ end2_w, const float* __restrict__ end2_b,
                            const float* __restrict__ fc1_w,  const float* __restrict__ fc1_b,
                            const float* __restrict__ fc2_w,  const float* __restrict__ fc2_b,
                            const float* __restrict__ fc3_w,  const float* __restrict__ fc3_b,
                            const float* __restrict__ fc4_w,  const float* __restrict__ fc4_b,
                            float* __restrict__ out)
{
    __shared__ float X[32][WCOLS];  // activations, data cols GUARD..GUARD+79
    __shared__ float R[32][WCOLS];  // dilated-conv "residual", cols GUARD..GUARD+95
    __shared__ float Y[32][WCOLS];  // f*g, cols GUARD..GUARD+79
    __shared__ float vA[256], vB[256];

    const int b    = blockIdx.x;       // batch element
    const int tid  = threadIdx.x;      // 128 threads = 4 wave32
    const int lane = tid & 31;
    const int wid  = tid >> 5;
    const int m    = lane & 15;        // M index within tile (A rows / C,D N slot)
    const int hi   = lane >> 4;        // half-wave select
    const int n    = m;                // N index within tile for B/C/D

    // ---- zero LDS (guards must be zero) ----
    for (int i = tid; i < 32 * WCOLS; i += 128) {
        ((float*)X)[i] = 0.f; ((float*)R)[i] = 0.f; ((float*)Y)[i] = 0.f;
    }
    __syncthreads();

    // ---- init: embedding lookup + 1x1 init conv over the trailing 80 tokens ----
    // X[ic][t] = init_b[ic] + sum_e init_w[ic][e] * emb[tok(t)][e]
    for (int job = tid; job < 32 * WIN; job += 128) {
        int t  = job % WIN;
        int ic = job / WIN;
        int tok = tokens[b * 8192 + (8192 - WIN) + t];
        const float* ev = emb + tok * 100;
        const float* iw = init_w + ic * 100;
        float acc = init_b[ic];
        for (int e = 0; e < 100; ++e) acc += iw[e] * ev[e];
        X[ic][GUARD + t] = acc;
    }
    __syncthreads();

    // ---- 16 dilated layers on the window ----
    for (int layer = 0; layer < 16; ++layer) {
        const int d = 1 << (layer & 3);
        const float* dw = dil_w  + layer * 2048;  // (32,32,2)
        const float* db = dil_b  + layer * 32;
        const float* fw = filt_w + layer * 2048;
        const float* fb = filt_b + layer * 32;
        const float* gw = gate_w + layer * 2048;
        const float* gb = gate_b + layer * 32;
        const float* rw = res_w  + layer * 1024;  // (32,32,1)
        const float* rb = res_b  + layer * 32;

        // Stage 1: R[t] = W0 @ X[t-(d+1)] + W1 @ X[t-1] + db, t = 0..95 (6 N-tiles)
        for (int job = wid; job < 12; job += 4) {
            int mt = job / 6, nt = job % 6;
            int oc0 = mt * 16, t0 = nt * 16;
            v16h a0 = a_frag(dw + 0, oc0, 64, 2, m, hi);
            v16h a1 = a_frag(dw + 1, oc0, 64, 2, m, hi);
            v16h b0 = b_frag(X, t0, -(d + 1), n, hi);
            v16h b1 = b_frag(X, t0, -1,       n, hi);
            v8f acc = {};
            acc = wmma_f16(a0, b0, acc);
            acc = wmma_f16(a1, b1, acc);
#pragma unroll
            for (int v = 0; v < 8; ++v) {
                int oc = oc0 + v + 8 * hi;
                R[oc][GUARD + t0 + n] = acc[v] + db[oc];
            }
        }
        __syncthreads();

        // Stage 2: Y[t] = tanh(F0@R[t]+F1@R[t+1]+fb) * sigmoid(G0@R[t]+G1@R[t+1]+gb)
        for (int job = wid; job < 10; job += 4) {
            int mt = job / 5, nt = job % 5;
            int oc0 = mt * 16, t0 = nt * 16;
            v16h af0 = a_frag(fw + 0, oc0, 64, 2, m, hi);
            v16h af1 = a_frag(fw + 1, oc0, 64, 2, m, hi);
            v16h ag0 = a_frag(gw + 0, oc0, 64, 2, m, hi);
            v16h ag1 = a_frag(gw + 1, oc0, 64, 2, m, hi);
            v16h b0  = b_frag(R, t0, 0, n, hi);
            v16h b1  = b_frag(R, t0, 1, n, hi);
            v8f aF = {}, aG = {};
            aF = wmma_f16(af0, b0, aF); aF = wmma_f16(af1, b1, aF);
            aG = wmma_f16(ag0, b0, aG); aG = wmma_f16(ag1, b1, aG);
#pragma unroll
            for (int v = 0; v < 8; ++v) {
                int oc = oc0 + v + 8 * hi;
                float fv = tanhf(aF[v] + fb[oc]);
                float gv = 1.f / (1.f + expf(-(aG[v] + gb[oc])));
                Y[oc][GUARD + t0 + n] = fv * gv;
            }
        }
        __syncthreads();

        // Stage 3: X[t] = Rw @ Y[t] + rb + R[t]   (not needed after layer 15)
        if (layer < 15) {
            for (int job = wid; job < 10; job += 4) {
                int mt = job / 5, nt = job % 5;
                int oc0 = mt * 16, t0 = nt * 16;
                v16h ar = a_frag(rw, oc0, 32, 1, m, hi);
                v16h by = b_frag(Y, t0, 0, n, hi);
                v8f acc = {};
                acc = wmma_f16(ar, by, acc);
#pragma unroll
                for (int v = 0; v < 8; ++v) {
                    int oc = oc0 + v + 8 * hi;
                    X[oc][GUARD + t0 + n] = acc[v] + rb[oc] + R[oc][GUARD + t0 + n];
                }
            }
            __syncthreads();
        }
    }

    // ---- head: only layer-15 skip at final position (t = 79, col GUARD+79) ----
    const float* skw = skip_w + 15 * 256 * 32;
    const float* skb = skip_b + 15 * 256;

    // skip vec, then relu
    for (int oc = tid; oc < 256; oc += 128) {
        float acc = skb[oc];
        for (int ic = 0; ic < 32; ++ic) acc += skw[oc * 32 + ic] * Y[ic][GUARD + 79];
        vA[oc] = fmaxf(acc, 0.f);
    }
    __syncthreads();

    // end1 (256x256) + relu
    for (int oc = tid; oc < 256; oc += 128) {
        float acc = end1_b[oc];
        for (int ic = 0; ic < 256; ++ic) acc += end1_w[oc * 256 + ic] * vA[ic];
        vB[oc] = fmaxf(acc, 0.f);
    }
    __syncthreads();

    // end2 (256x256), no relu
    for (int oc = tid; oc < 256; oc += 128) {
        float acc = end2_b[oc];
        for (int ic = 0; ic < 256; ++ic) acc += end2_w[oc * 256 + ic] * vB[ic];
        vA[oc] = acc;
    }
    __syncthreads();

    // fc1 (128x256) + relu
    for (int oc = tid; oc < 128; oc += 128) {
        float acc = fc1_b[oc];
        for (int ic = 0; ic < 256; ++ic) acc += fc1_w[oc * 256 + ic] * vA[ic];
        vB[oc] = fmaxf(acc, 0.f);
    }
    __syncthreads();

    // fc2 (128x128) + relu
    for (int oc = tid; oc < 128; oc += 128) {
        float acc = fc2_b[oc];
        for (int ic = 0; ic < 128; ++ic) acc += fc2_w[oc * 128 + ic] * vB[ic];
        vA[oc] = fmaxf(acc, 0.f);
    }
    __syncthreads();

    // fc3 (64x128) + relu
    for (int oc = tid; oc < 64; oc += 128) {
        float acc = fc3_b[oc];
        for (int ic = 0; ic < 128; ++ic) acc += fc3_w[oc * 128 + ic] * vA[ic];
        vB[oc] = fmaxf(acc, 0.f);
    }
    __syncthreads();

    // fc4 (256x64) -> out[b][0..255]
    for (int oc = tid; oc < 256; oc += 128) {
        float acc = fc4_b[oc];
        for (int ic = 0; ic < 64; ++ic) acc += fc4_w[oc * 64 + ic] * vB[ic];
        out[b * 256 + oc] = acc;
    }
}

extern "C" void kernel_launch(void* const* d_in, const int* in_sizes, int n_in,
                              void* d_out, int out_size, void* d_ws, size_t ws_size,
                              hipStream_t stream) {
    (void)in_sizes; (void)n_in; (void)out_size; (void)d_ws; (void)ws_size;
    const int*   tokens = (const int*)d_in[0];
    const float* emb    = (const float*)d_in[1];
    const float* init_w = (const float*)d_in[2];
    const float* init_b = (const float*)d_in[3];
    const float* dil_w  = (const float*)d_in[4];
    const float* dil_b  = (const float*)d_in[5];
    const float* filt_w = (const float*)d_in[6];
    const float* filt_b = (const float*)d_in[7];
    const float* gate_w = (const float*)d_in[8];
    const float* gate_b = (const float*)d_in[9];
    const float* res_w  = (const float*)d_in[10];
    const float* res_b  = (const float*)d_in[11];
    const float* skip_w = (const float*)d_in[12];
    const float* skip_b = (const float*)d_in[13];
    const float* end1_w = (const float*)d_in[14];
    const float* end1_b = (const float*)d_in[15];
    const float* end2_w = (const float*)d_in[16];
    const float* end2_b = (const float*)d_in[17];
    const float* fc1_w  = (const float*)d_in[18];
    const float* fc1_b  = (const float*)d_in[19];
    const float* fc2_w  = (const float*)d_in[20];
    const float* fc2_b  = (const float*)d_in[21];
    const float* fc3_w  = (const float*)d_in[22];
    const float* fc3_b  = (const float*)d_in[23];
    const float* fc4_w  = (const float*)d_in[24];
    const float* fc4_b  = (const float*)d_in[25];

    wavenet_gfx1250_kernel<<<16, 128, 0, stream>>>(
        tokens, emb, init_w, init_b, dil_w, dil_b, filt_w, filt_b,
        gate_w, gate_b, res_w, res_b, skip_w, skip_b, end1_w, end1_b,
        end2_w, end2_b, fc1_w, fc1_b, fc2_w, fc2_b, fc3_w, fc3_b,
        fc4_w, fc4_b, (float*)d_out);
}
